// MultiLayer_Model_56100862820650
// MI455X (gfx1250) — matrix operand order repaired and checked by reading, not compiled, not run on hardware
//
#include <hip/hip_runtime.h>

// ---------------- problem constants ----------------
constexpr int LAYERS = 5;
constexpr int BATCH  = 64;
constexpr int TSTEPS = 256;
constexpr int HID    = 1024;
constexpr int NG     = 4 * HID;   // gate-packed N dimension (n = col*4 + gate)
constexpr int LP     = 40;        // padded LDS row stride (bf16 elems): 80B -> conflict-free b128 reads

typedef __bf16 v16bf __attribute__((ext_vector_type(16)));
typedef __bf16 v8bf  __attribute__((ext_vector_type(8)));
typedef float  v8f   __attribute__((ext_vector_type(8)));

static __device__ __forceinline__ v16bf combine16(v8bf lo, v8bf hi) {
    v16bf r;
#pragma unroll
    for (int i = 0; i < 8; ++i) { r[i] = lo[i]; r[i + 8] = hi[i]; }
    return r;
}

static __device__ __forceinline__ float sigmoidf_(float x) {
    return 1.0f / (1.0f + __expf(-x));
}

// ---------------- packing kernels ----------------
// wT[l][n][k] layout, n = col*4 + gate (gate: 0=i,1=f,2=c,3=o)
// wxT from rows 0..H-1 of W (x part), whT from rows H..2H-1 (h part).
__global__ void pack_weights_kernel(const float* __restrict__ Wi, const float* __restrict__ Wf,
                                    const float* __restrict__ Wc, const float* __restrict__ Wo,
                                    __bf16* __restrict__ wxT, __bf16* __restrict__ whT) {
    size_t gid = (size_t)blockIdx.x * 256 + threadIdx.x;
    size_t total = (size_t)LAYERS * NG * HID;
    if (gid >= total) return;
    int k = (int)(gid % HID);
    int n = (int)((gid / HID) % NG);
    int l = (int)(gid / ((size_t)HID * NG));
    int col = n >> 2, g = n & 3;
    const float* W = (g == 0) ? Wi : (g == 1) ? Wf : (g == 2) ? Wc : Wo;
    size_t base = (size_t)l * 2 * HID * HID;
    wxT[gid] = (__bf16)W[base + (size_t)k * HID + col];
    whT[gid] = (__bf16)W[base + (size_t)(HID + k) * HID + col];
}

__global__ void pack_bias_kernel(const float* __restrict__ bi, const float* __restrict__ bff,
                                 const float* __restrict__ bc, const float* __restrict__ bo,
                                 float* __restrict__ biasP) {
    int gid = blockIdx.x * 256 + threadIdx.x;
    if (gid >= LAYERS * NG) return;
    int n = gid % NG, l = gid / NG, col = n >> 2, g = n & 3;
    const float* b = (g == 0) ? bi : (g == 1) ? bff : (g == 2) ? bc : bo;
    biasP[gid] = b[l * HID + col];
}

// x[B][T][H] f32  ->  seq[T][B][H] bf16
__global__ void pack_input_kernel(const float* __restrict__ x, __bf16* __restrict__ seq) {
    size_t gid = (size_t)blockIdx.x * 256 + threadIdx.x;   // == (t*B + b)*H + h
    int h = (int)(gid % HID);
    int b = (int)((gid / HID) % BATCH);
    int t = (int)(gid / ((size_t)HID * BATCH));
    seq[gid] = (__bf16)x[((size_t)b * TSTEPS + t) * HID + h];
}

__global__ void init_layer_kernel(__bf16* __restrict__ slot0, float* __restrict__ cstate) {
    int i = blockIdx.x * 256 + threadIdx.x;
    if (i < BATCH * HID) { slot0[i] = (__bf16)0.0f; cstate[i] = 0.0f; }
}

// ---------------- big parallel GEMM: preact = seq @ WxT^T + bias ----------------
// Block tile: 64(M) x 128(N). 8 waves, each wave = 1 m-tile x 4 n-tiles (A frag reused 4x).
// seq: [T*B][H] bf16, wT: [NG][H] bf16 (N-major K), preact: [T*B][NG] f32
__global__ __launch_bounds__(256)
void gemm_x_kernel(const __bf16* __restrict__ seq, const __bf16* __restrict__ wT,
                   const float* __restrict__ biasP, float* __restrict__ preact) {
    __shared__ __align__(16) __bf16 sA[64 * LP];
    __shared__ __align__(16) __bf16 sB[128 * LP];

    const int tid = threadIdx.x, wave = tid >> 5, lane = tid & 31;
    const int half = lane >> 4, lrow = lane & 15;
    const int mt = wave & 3, nh = wave >> 2;          // m-tile 0..3, n-half 0..1 (4 n-tiles each)
    const int n0 = blockIdx.x * 128;
    const size_t m0 = (size_t)blockIdx.y * 64;
    const int sr = tid >> 2, sk = (tid & 3) * 8;      // staging: row, k-quad

    v8f acc[4];
#pragma unroll
    for (int j = 0; j < 4; ++j) acc[j] = (v8f){0, 0, 0, 0, 0, 0, 0, 0};

#pragma unroll 2
    for (int kc = 0; kc < HID; kc += 32) {
        __syncthreads();
        *(v8bf*)&sA[sr * LP + sk] = *(const v8bf*)&seq[(m0 + sr) * HID + kc + sk];
#pragma unroll
        for (int j = 0; j < 2; ++j) {
            int br = sr + 64 * j;
            *(v8bf*)&sB[br * LP + sk] = *(const v8bf*)&wT[(size_t)(n0 + br) * HID + kc + sk];
        }
        if (kc + 32 < HID)
            __builtin_prefetch(&wT[(size_t)(n0 + sr) * HID + kc + 32 + sk], 0, 3);
        __syncthreads();

        // Load A fragment and ALL four B fragments first (10 ds_load_b128 in flight),
        // then issue the WMMA chain -- avoids load->wait0->wmma serialization.
        const __bf16* ap = &sA[(mt * 16 + lrow) * LP];
        v16bf a = combine16(*(const v8bf*)&ap[8 * half], *(const v8bf*)&ap[16 + 8 * half]);
        v16bf bfr[4];
#pragma unroll
        for (int j = 0; j < 4; ++j) {
            const __bf16* bp = &sB[((nh * 4 + j) * 16 + lrow) * LP];
            bfr[j] = combine16(*(const v8bf*)&bp[8 * half], *(const v8bf*)&bp[16 + 8 * half]);
        }
#pragma unroll
        for (int j = 0; j < 4; ++j)
            acc[j] = __builtin_amdgcn_wmma_f32_16x16x32_bf16(false, a, false, bfr[j], (short)0, acc[j], false, false);
    }

#pragma unroll
    for (int j = 0; j < 4; ++j) {
        const int n = n0 + (nh * 4 + j) * 16 + lrow;
        const float bn = biasP[n];
#pragma unroll
        for (int r = 0; r < 8; ++r) {
            size_t row = m0 + mt * 16 + half * 8 + r;
            preact[row * NG + n] = acc[j][r] + bn;
        }
    }
}

// ---------------- recurrent step: gates = preact[t] + h_{t-1} @ WhT^T ; fused LSTM cell ----------------
// Block tile: 64(M=batch) x 32(N). 8 waves, 1 WMMA tile each -> 128 blocks for occupancy on the
// latency-bound sequential path. N=32 still holds all 4 gates for 8 columns (gate-interleaved N).
__global__ __launch_bounds__(256)
void lstm_step_kernel(const float* __restrict__ preact_t,   // [B][NG] f32 (x-part + bias)
                      const __bf16* __restrict__ hprev,     // [B][H] bf16
                      const __bf16* __restrict__ whT,       // [NG][H] bf16
                      float* __restrict__ cstate,           // [B][H] f32
                      __bf16* __restrict__ hout) {          // [B][H] bf16
    __shared__ __align__(16) __bf16 sA[64 * LP];
    __shared__ __align__(16) __bf16 sB[32 * LP];
    __shared__ float sD[64 * 32];

    const int tid = threadIdx.x, wave = tid >> 5, lane = tid & 31;
    const int half = lane >> 4, lrow = lane & 15;
    const int mt = wave & 3, nt = wave >> 2;          // m-tile 0..3, n-tile 0..1
    const int n0 = blockIdx.x * 32;
    const int sr = tid >> 2, sk = (tid & 3) * 8;

    const int nIdx = n0 + nt * 16 + lrow;

    // Initialize accumulator from precomputed x-part (C-matrix layout: M = 8*half + r)
    v8f acc;
#pragma unroll
    for (int r = 0; r < 8; ++r) {
        int brow = mt * 16 + half * 8 + r;
        acc[r] = preact_t[(size_t)brow * NG + nIdx];
    }

#pragma unroll 2
    for (int kc = 0; kc < HID; kc += 32) {
        __syncthreads();
        *(v8bf*)&sA[sr * LP + sk] = *(const v8bf*)&hprev[(size_t)sr * HID + kc + sk];
        if (tid < 128)
            *(v8bf*)&sB[sr * LP + sk] = *(const v8bf*)&whT[(size_t)(n0 + sr) * HID + kc + sk];
        if (kc + 32 < HID)
            __builtin_prefetch(&whT[(size_t)(n0 + (sr & 31)) * HID + kc + 32 + sk], 0, 3);
        __syncthreads();

        const __bf16* ap = &sA[(mt * 16 + lrow) * LP];
        v16bf a = combine16(*(const v8bf*)&ap[8 * half], *(const v8bf*)&ap[16 + 8 * half]);
        const __bf16* bp = &sB[(nt * 16 + lrow) * LP];
        v16bf b = combine16(*(const v8bf*)&bp[8 * half], *(const v8bf*)&bp[16 + 8 * half]);
        acc = __builtin_amdgcn_wmma_f32_16x16x32_bf16(false, a, false, b, (short)0, acc, false, false);
    }

    // Dump gate pre-activations to LDS: block owns all 4 gates for 8 columns x 64 batch rows
    __syncthreads();
#pragma unroll
    for (int r = 0; r < 8; ++r) {
        int brow = mt * 16 + half * 8 + r;
        sD[brow * 32 + nt * 16 + lrow] = acc[r];
    }
    __syncthreads();

    // Fused LSTM cell update: 64 batch rows x 8 columns = 512 elements, 2 per thread
#pragma unroll
    for (int i = 0; i < 2; ++i) {
        int idx = tid + i * 256;
        int b = idx >> 3;
        int cl = idx & 7;
        float gi = sD[b * 32 + cl * 4 + 0];
        float gf = sD[b * 32 + cl * 4 + 1];
        float gc = sD[b * 32 + cl * 4 + 2];
        float go = sD[b * 32 + cl * 4 + 3];
        int col = (n0 >> 2) + cl;
        float c_old = cstate[b * HID + col];
        float c_new = c_old * sigmoidf_(gf) + sigmoidf_(gi) * tanhf(gc);
        cstate[b * HID + col] = c_new;
        hout[b * HID + col] = (__bf16)(sigmoidf_(go) * tanhf(c_new));
    }
}

// ---------------- final FC: out[b] = last[b,:] . Wfc + bfc ----------------
__global__ __launch_bounds__(256)
void fc_kernel(const __bf16* __restrict__ last, const float* __restrict__ Wfc,
               const float* __restrict__ bfc, float* __restrict__ out) {
    __shared__ float red[256];
    int b = blockIdx.x;
    float s = 0.0f;
    for (int h = threadIdx.x; h < HID; h += 256)
        s += (float)last[(size_t)b * HID + h] * Wfc[h];
    red[threadIdx.x] = s;
    __syncthreads();
    for (int off = 128; off > 0; off >>= 1) {
        if (threadIdx.x < off) red[threadIdx.x] += red[threadIdx.x + off];
        __syncthreads();
    }
    if (threadIdx.x == 0) out[b] = red[0] + bfc[0];
}

// ---------------- host ----------------
extern "C" void kernel_launch(void* const* d_in, const int* in_sizes, int n_in,
                              void* d_out, int out_size, void* d_ws, size_t ws_size,
                              hipStream_t stream) {
    (void)in_sizes; (void)n_in; (void)out_size; (void)ws_size;
    const float* x   = (const float*)d_in[0];
    const float* Wi  = (const float*)d_in[1];
    const float* bi  = (const float*)d_in[2];
    const float* Wf  = (const float*)d_in[3];
    const float* bff = (const float*)d_in[4];
    const float* Wc  = (const float*)d_in[5];
    const float* bc  = (const float*)d_in[6];
    const float* Wo  = (const float*)d_in[7];
    const float* bo  = (const float*)d_in[8];
    const float* Wfc = (const float*)d_in[9];
    const float* bfc = (const float*)d_in[10];
    float* out = (float*)d_out;

    char* ws = (char*)d_ws;
    size_t off = 0;
    auto alloc = [&](size_t bytes) -> char* {
        char* p = ws + off;
        off += (bytes + 255) & ~(size_t)255;
        return p;
    };

    const size_t WT_ELEMS  = (size_t)LAYERS * NG * HID;          // per weight-half
    const size_t BUF_ELEMS = (size_t)(TSTEPS + 1) * BATCH * HID; // hidden seq (+ zero slot)
    const size_t PRE_ELEMS = (size_t)TSTEPS * BATCH * NG;

    __bf16* wxT    = (__bf16*)alloc(WT_ELEMS * sizeof(__bf16));
    __bf16* whT    = (__bf16*)alloc(WT_ELEMS * sizeof(__bf16));
    float*  biasP  = (float*) alloc((size_t)LAYERS * NG * sizeof(float));
    __bf16* bufA   = (__bf16*)alloc(BUF_ELEMS * sizeof(__bf16));
    __bf16* bufB   = (__bf16*)alloc(BUF_ELEMS * sizeof(__bf16));
    float*  cstate = (float*) alloc((size_t)BATCH * HID * sizeof(float));
    float*  preact = (float*) alloc(PRE_ELEMS * sizeof(float));

    // 1. pack weights / bias / input
    pack_weights_kernel<<<(unsigned)((WT_ELEMS + 255) / 256), 256, 0, stream>>>(
        Wi, Wf, Wc, Wo, wxT, whT);
    pack_bias_kernel<<<(LAYERS * NG + 255) / 256, 256, 0, stream>>>(bi, bff, bc, bo, biasP);
    pack_input_kernel<<<(unsigned)(((size_t)TSTEPS * BATCH * HID) / 256), 256, 0, stream>>>(
        x, bufA + (size_t)BATCH * HID);

    // 2. layers
    __bf16* bin = bufA;
    __bf16* bout = bufB;
    for (int l = 0; l < LAYERS; ++l) {
        init_layer_kernel<<<(BATCH * HID) / 256, 256, 0, stream>>>(bout, cstate);
        // parallel x-part GEMM for all timesteps
        gemm_x_kernel<<<dim3(NG / 128, (TSTEPS * BATCH) / 64), 256, 0, stream>>>(
            bin + (size_t)BATCH * HID, wxT + (size_t)l * NG * HID,
            biasP + (size_t)l * NG, preact);
        // sequential recurrent steps
        for (int t = 0; t < TSTEPS; ++t) {
            lstm_step_kernel<<<NG / 32, 256, 0, stream>>>(
                preact + (size_t)t * BATCH * NG,
                bout + (size_t)t * BATCH * HID,        // h_{t-1} (slot t; slot 0 zeroed)
                whT + (size_t)l * NG * HID,
                cstate,
                bout + (size_t)(t + 1) * BATCH * HID); // h_t -> slot t+1
        }
        __bf16* tmp = bin; bin = bout; bout = tmp;
    }

    // 3. FC on last hidden state (bin now points at last layer's output buffer)
    fc_kernel<<<BATCH, 256, 0, stream>>>(bin + (size_t)TSTEPS * BATCH * HID, Wfc, bfc, out);
}
